// VGNN_71236327572107
// MI455X (gfx1250) — matrix operand order repaired and checked by reading, not compiled, not run on hardware
//
#include <hip/hip_runtime.h>
#include <hip/hip_bf16.h>
#include <math.h>

// ---------------------------------------------------------------------------
// Problem constants (from the reference)
// ---------------------------------------------------------------------------
#define TT   32
#define NN   4096
#define FEAT 64
#define NHID 128
#define HR   128
#define HS   64
#define G3H  384      // 3*HR
#define ALPHA 0.2f
#define NEGV (-9.0e10f)

// ---------------------------------------------------------------------------
// WMMA types (CDNA5 / gfx1250, wave32)
// ---------------------------------------------------------------------------
typedef __attribute__((ext_vector_type(16))) __bf16 v16bf;
typedef __attribute__((ext_vector_type(8)))  float  v8f;

__device__ __forceinline__ float sigmoidf_(float x) {
    return 1.0f / (1.0f + __expf(-x));
}

// ---------------------------------------------------------------------------
// Generic bf16-WMMA GEMM:  C[M,N] = A[M,K] * op(B) + bias
//   op(B) = B[K,N] (transB=0)  or  B[N,K]^T (transB=1)
//
// REQUIREMENTS (all call sites satisfy these; no bounds checks on purpose —
// guarded loads scalarize the LDS staging into per-element exec-mask branches):
//   M % 128 == 0, N % 64 == 0, K % 32 == 0, lda/ldb % 4 == 0.
//
// Tile: 128x64 per 256-thread block (8 waves), K stepped by 32.
// Wave w owns rows [w*16, w*16+16) x all 64 cols: 1 A-frag, 4 B-frags,
// 4 accumulators, 4 WMMA per K-chunk.
//
// Fragment layouts (ISA 7.12.2, wave32):
//   A 16x32 bf16 : lane L row M=L%16; elems 0..7 = K (L/16)*8..+7, 8..15 = +16.
//   B 32x16 bf16 : lane L col N=L%16; elems 0..7 = K (L/16)*8..+7, 8..15 = +16.
//   C/D 16x16 f32: lane L col N=L%16; acc[r] = row (L/16)*8 + r.
// ---------------------------------------------------------------------------
#define GTM 128
#define GTN 64
#define GTK 32
#define LROW 40                 // 32 + 8 pad -> 80B rows (16B-aligned)

__global__ __launch_bounds__(256)
void gemm_bf16_wmma_kernel(const float* __restrict__ A, int lda,
                           const float* __restrict__ B, int ldb, int transB,
                           float* __restrict__ C, int ldc,
                           int M, int N, int K,
                           const float* __restrict__ bias)
{
    __shared__ __bf16 As[GTM][LROW];    // [m][k]
    __shared__ __bf16 BsT[GTN][LROW];   // [n][k]  (K contiguous)

    const int tid   = threadIdx.x;
    const int wave  = tid >> 5;
    const int lane  = tid & 31;
    const int wgM   = blockIdx.y * GTM;
    const int wgN   = blockIdx.x * GTN;
    const int subM  = wave * 16;            // 0..112
    const int lrow  = lane & 15;
    const int lhalf = lane >> 4;

    // staging decomposition: 4 threads per 32-wide row segment
    const int srow = tid >> 2;              // 0..63
    const int scol = (tid & 3) * 8;         // 0,8,16,24

    v8f acc[4];
    #pragma unroll
    for (int i = 0; i < 4; ++i) acc[i] = (v8f){0.f,0.f,0.f,0.f,0.f,0.f,0.f,0.f};

    for (int k0 = 0; k0 < K; k0 += GTK) {
        // ---- stage A tile 128x32: 2 x (global b128 pair -> 8 bf16 -> ds b128) ----
        #pragma unroll
        for (int h = 0; h < 2; ++h) {
            const float* ga = A + (size_t)(wgM + srow + h * 64) * lda + (k0 + scol);
            float4 p0 = *(const float4*)ga;
            float4 p1 = *(const float4*)(ga + 4);
            union { __bf16 hv[8]; uint4 u; } pk;
            pk.hv[0] = (__bf16)p0.x; pk.hv[1] = (__bf16)p0.y;
            pk.hv[2] = (__bf16)p0.z; pk.hv[3] = (__bf16)p0.w;
            pk.hv[4] = (__bf16)p1.x; pk.hv[5] = (__bf16)p1.y;
            pk.hv[6] = (__bf16)p1.z; pk.hv[7] = (__bf16)p1.w;
            *(uint4*)&As[srow + h * 64][scol] = pk.u;
        }
        // ---- stage B tile 32x64 into BsT[n][k] ----
        if (transB) {
            // B[N,K]: K-contiguous, same vector path as A (64 rows x 4 thr/row)
            const float* gb = B + (size_t)(wgN + srow) * ldb + (k0 + scol);
            float4 p0 = *(const float4*)gb;
            float4 p1 = *(const float4*)(gb + 4);
            union { __bf16 hv[8]; uint4 u; } pk;
            pk.hv[0] = (__bf16)p0.x; pk.hv[1] = (__bf16)p0.y;
            pk.hv[2] = (__bf16)p0.z; pk.hv[3] = (__bf16)p0.w;
            pk.hv[4] = (__bf16)p1.x; pk.hv[5] = (__bf16)p1.y;
            pk.hv[6] = (__bf16)p1.z; pk.hv[7] = (__bf16)p1.w;
            *(uint4*)&BsT[srow][scol] = pk.u;
        } else {
            // B[K,N]: gather k-pairs per fixed n; pack 2 bf16 -> b32 store
            const int bn = tid & 63;
            const int bk = (tid >> 6) * 2;  // 0,2,4,6
            #pragma unroll
            for (int kk = 0; kk < 4; ++kk) {
                int k = bk + kk * 8;
                float x0 = B[(size_t)(k0 + k)     * ldb + (wgN + bn)];
                float x1 = B[(size_t)(k0 + k + 1) * ldb + (wgN + bn)];
                union { __bf16 hv[2]; unsigned u; } pk;
                pk.hv[0] = (__bf16)x0; pk.hv[1] = (__bf16)x1;
                *(unsigned*)&BsT[bn][k] = pk.u;
            }
        }
        // prefetch next A tile (global_prefetch_b8), uniform scalar guard
        if (k0 + GTK < K) {
            __builtin_prefetch(A + (size_t)(wgM + srow) * lda + (k0 + GTK + scol), 0, 1);
        }
        __syncthreads();

        // ---- fragments from LDS (ds_load_b128 x2 each) ----
        union { v16bf v; uint4 u[2]; } af;
        const __bf16* ap = &As[subM + lrow][lhalf * 8];
        af.u[0] = *(const uint4*)(ap);
        af.u[1] = *(const uint4*)(ap + 16);

        #pragma unroll
        for (int ns = 0; ns < 4; ++ns) {
            union { v16bf v; uint4 u[2]; } bfr;
            const __bf16* bp = &BsT[ns * 16 + lrow][lhalf * 8];
            bfr.u[0] = *(const uint4*)(bp);
            bfr.u[1] = *(const uint4*)(bp + 16);
            acc[ns] = __builtin_amdgcn_wmma_f32_16x16x32_bf16(
                false, af.v, false, bfr.v, (short)0, acc[ns], false, false);
        }
        __syncthreads();
    }

    // ---- store C (+bias) ----
    const int rbase = wgM + subM + lhalf * 8;
    #pragma unroll
    for (int ns = 0; ns < 4; ++ns) {
        const int cn = wgN + ns * 16 + lrow;
        const float bv = bias ? bias[cn] : 0.0f;
        #pragma unroll
        for (int r = 0; r < 8; ++r) {
            C[(size_t)(rbase + r) * ldc + cn] = acc[ns][r] + bv;
        }
    }
}

// ---------------------------------------------------------------------------
// Small prep: W' = W_fc @ Wih^T  [64,384];  b' = bih + b_fc @ Wih^T  [384];
//             v1 = W_imp @ a1, v2 = W_imp @ a2  [128] each.
// ---------------------------------------------------------------------------
__global__ void prep_kernel(const float* __restrict__ W_fc, const float* __restrict__ b_fc,
                            const float* __restrict__ Wih,  const float* __restrict__ bih,
                            const float* __restrict__ W_imp,
                            const float* __restrict__ a1,   const float* __restrict__ a2,
                            float* __restrict__ Wp, float* __restrict__ bprime,
                            float* __restrict__ v1, float* __restrict__ v2)
{
    int idx = blockIdx.x * blockDim.x + threadIdx.x;
    const int NWP = FEAT * G3H;                     // 24576
    if (idx < NWP) {
        int f = idx / G3H, j = idx % G3H;
        float s = 0.0f;
        for (int k = 0; k < NHID; ++k) s += W_fc[f * NHID + k] * Wih[j * NHID + k];
        Wp[idx] = s;
    } else if (idx < NWP + G3H) {
        int j = idx - NWP;
        float s = bih[j];
        for (int k = 0; k < NHID; ++k) s += b_fc[k] * Wih[j * NHID + k];
        bprime[j] = s;
    } else if (idx < NWP + G3H + HR) {
        int k = idx - (NWP + G3H);
        float s = 0.0f;
        for (int c = 0; c < NHID; ++c) s += W_imp[k * NHID + c] * a1[c];
        v1[k] = s;
    } else if (idx < NWP + G3H + 2 * HR) {
        int k = idx - (NWP + G3H + HR);
        float s = 0.0f;
        for (int c = 0; c < NHID; ++c) s += W_imp[k * NHID + c] * a2[c];
        v2[k] = s;
    }
}

// ---------------------------------------------------------------------------
// GRU pointwise gate update:  h_out = (1-z)*n + z*h_in
// ---------------------------------------------------------------------------
__global__ void gru_update_kernel(const float* __restrict__ Gi, const float* __restrict__ Gh,
                                  const float* __restrict__ h_in, float* __restrict__ h_out)
{
    int idx = blockIdx.x * 256 + threadIdx.x;          // < N*HR
    int n = idx >> 7, c = idx & (HR - 1);
    const float* gi = Gi + (size_t)n * G3H;
    const float* gh = Gh + (size_t)n * G3H;
    float r  = sigmoidf_(gi[c]          + gh[c]);
    float z  = sigmoidf_(gi[HR + c]     + gh[HR + c]);
    float nn = tanhf   (gi[2 * HR + c] + r * gh[2 * HR + c]);
    h_out[idx] = (1.0f - z) * nn + z * h_in[idx];
}

// f[n] = h[n] . v1 ; g[n] = h[n] . v2
__global__ void fg_kernel(const float* __restrict__ h, const float* __restrict__ v1,
                          const float* __restrict__ v2, float* __restrict__ f,
                          float* __restrict__ g)
{
    int n = blockIdx.x * 256 + threadIdx.x;
    if (n >= NN) return;
    float sf = 0.0f, sg = 0.0f;
    const float* hp = h + (size_t)n * HR;
    for (int k = 0; k < HR; ++k) { float hv = hp[k]; sf += hv * v1[k]; sg += hv * v2[k]; }
    f[n] = sf; g[n] = sg;
}

// relation = leaky(f[i]+g[j]) + A_Ind*S ; write implicit to d_out; mask into R
__global__ void relation_kernel(const float* __restrict__ f, const float* __restrict__ g,
                                const float* __restrict__ A_Ind,
                                float* __restrict__ R, float* __restrict__ imp_out)
{
    size_t idx = (size_t)blockIdx.x * 256 + threadIdx.x;   // < N*N
    int i = (int)(idx >> 12), j = (int)(idx & (NN - 1));
    float e   = f[i] + g[j];
    float imp = e > 0.0f ? e : ALPHA * e;
    imp_out[idx] = imp;
    float rel = imp + A_Ind[idx] * R[idx];
    R[idx] = rel > 0.0f ? rel : NEGV;
}

// in-place row softmax over R [N, N]
__global__ void softmax_row_kernel(float* __restrict__ R)
{
    __shared__ float red[256];
    const int tid = threadIdx.x;
    float* x = R + (size_t)blockIdx.x * NN;

    float m = -3.4e38f;
    for (int j = tid; j < NN; j += 256) m = fmaxf(m, x[j]);
    red[tid] = m; __syncthreads();
    for (int s = 128; s > 0; s >>= 1) {
        if (tid < s) red[tid] = fmaxf(red[tid], red[tid + s]);
        __syncthreads();
    }
    m = red[0]; __syncthreads();

    float sum = 0.0f;
    for (int j = tid; j < NN; j += 256) { float e = __expf(x[j] - m); x[j] = e; sum += e; }
    red[tid] = sum; __syncthreads();
    for (int s = 128; s > 0; s >>= 1) {
        if (tid < s) red[tid] += red[tid + s];
        __syncthreads();
    }
    float inv = 1.0f / red[0];
    for (int j = tid; j < NN; j += 256) x[j] *= inv;
}

// gate_Mh = Mh * sigmoid(Att)
__global__ void gate_kernel(const float* __restrict__ Mh, const float* __restrict__ Att,
                            float* __restrict__ gateMh)
{
    int idx = blockIdx.x * 256 + threadIdx.x;              // < N*HS
    gateMh[idx] = Mh[idx] * sigmoidf_(Att[idx]);
}

// head: out[n] = elu([Hs[n], h[n]] @ W1 + b1) @ W2 + b2
__global__ void head_kernel(const float* __restrict__ Hs, const float* __restrict__ h,
                            const float* __restrict__ W1, const float* __restrict__ b1,
                            const float* __restrict__ W2, const float* __restrict__ b2,
                            float* __restrict__ out)
{
    __shared__ float sW1[(HS + HR) * 16];
    __shared__ float sb1[16], sW2[16];
    const int tid = threadIdx.x;
    for (int i = tid; i < (HS + HR) * 16; i += 256) sW1[i] = W1[i];
    if (tid < 16) { sb1[tid] = b1[tid]; sW2[tid] = W2[tid]; }
    __syncthreads();

    int n = blockIdx.x * 256 + tid;
    if (n >= NN) return;
    float t[16];
    #pragma unroll
    for (int j = 0; j < 16; ++j) t[j] = sb1[j];
    const float* hs = Hs + (size_t)n * HS;
    for (int k = 0; k < HS; ++k) {
        float v = hs[k];
        #pragma unroll
        for (int j = 0; j < 16; ++j) t[j] += v * sW1[k * 16 + j];
    }
    const float* hp = h + (size_t)n * HR;
    for (int k = 0; k < HR; ++k) {
        float v = hp[k];
        #pragma unroll
        for (int j = 0; j < 16; ++j) t[j] += v * sW1[(HS + k) * 16 + j];
    }
    float acc = b2[0];
    #pragma unroll
    for (int j = 0; j < 16; ++j) {
        float e = t[j] > 0.0f ? t[j] : (__expf(t[j]) - 1.0f);
        acc += e * sW2[j];
    }
    out[n] = acc;
}

// ---------------------------------------------------------------------------
// Host-side orchestration
// ---------------------------------------------------------------------------
static inline void launch_gemm(const float* A, int lda, const float* B, int ldb,
                               int transB, float* C, int ldc, int M, int N, int K,
                               const float* bias, hipStream_t stream)
{
    dim3 grid((N + GTN - 1) / GTN, (M + GTM - 1) / GTM);
    gemm_bf16_wmma_kernel<<<grid, 256, 0, stream>>>(A, lda, B, ldb, transB,
                                                    C, ldc, M, N, K, bias);
}

extern "C" void kernel_launch(void* const* d_in, const int* in_sizes, int n_in,
                              void* d_out, int out_size, void* d_ws, size_t ws_size,
                              hipStream_t stream)
{
    (void)in_sizes; (void)n_in; (void)out_size; (void)ws_size;

    const float* A_Ind   = (const float*)d_in[0];   // [N,N]
    const float* inputd  = (const float*)d_in[1];   // [T,N,FEAT]
    const float* W_fc    = (const float*)d_in[2];   // [FEAT,NHID]
    const float* b_fc    = (const float*)d_in[3];   // [NHID]
    const float* Wih     = (const float*)d_in[4];   // [3H,NHID]
    const float* Whh     = (const float*)d_in[5];   // [3H,HR]
    const float* bih     = (const float*)d_in[6];   // [3H]
    const float* bhh     = (const float*)d_in[7];   // [3H]
    const float* W_imp   = (const float*)d_in[8];   // [HR,NHID]
    const float* a1      = (const float*)d_in[9];   // [NHID,1]
    const float* a2      = (const float*)d_in[10];  // [NHID,1]
    const float* W_exp   = (const float*)d_in[11];  // [HR,NHID]
    const float* W_att   = (const float*)d_in[12];  // [HR,HS]
    const float* b_att   = (const float*)d_in[13];  // [HS]
    const float* graph_W = (const float*)d_in[14];  // [HR,HS]
    const float* W1      = (const float*)d_in[15];  // [HR+HS,16]
    const float* b1      = (const float*)d_in[16];  // [16]
    const float* W2      = (const float*)d_in[17];  // [16,1]
    const float* b2      = (const float*)d_in[18];  // [1]

    float* out_H   = (float*)d_out;                 // [N]
    float* out_imp = out_H + NN;                    // [N,N]
    float* out_reg = out_imp + (size_t)NN * NN;     // [1]

    // ---- carve scratch out of d_ws ----
    char* wsp = (char*)d_ws;
    auto alloc = [&](size_t nfloats) -> float* {
        float* p = (float*)wsp;
        wsp += ((nfloats * sizeof(float) + 255) / 256) * 256;
        return p;
    };
    float* Wp     = alloc((size_t)FEAT * G3H);      // fused W_fc@Wih^T
    float* bprime = alloc(G3H);
    float* v1     = alloc(HR);
    float* v2     = alloc(HR);
    float* fvec   = alloc(NN);
    float* gvec   = alloc(NN);
    float* Gi     = alloc((size_t)NN * G3H);
    float* Gh     = alloc((size_t)NN * G3H);
    float* hA     = alloc((size_t)NN * HR);
    float* hB     = alloc((size_t)NN * HR);
    float* He     = alloc((size_t)NN * NHID);
    float* Mh     = alloc((size_t)NN * HS);
    float* Att    = alloc((size_t)NN * HS);
    float* gateMh = alloc((size_t)NN * HS);
    float* Hs     = alloc((size_t)NN * HS);
    float* R      = alloc((size_t)NN * NN);         // relation / softmax A (64 MB)

    // ---- 1) small precompute (linear-layer fusion) ----
    {
        int total = FEAT * G3H + G3H + 2 * HR;
        prep_kernel<<<(total + 255) / 256, 256, 0, stream>>>(
            W_fc, b_fc, Wih, bih, W_imp, a1, a2, Wp, bprime, v1, v2);
    }

    // ---- 2) GRU scan over T=32 steps ----
    hipMemsetAsync(hA, 0, (size_t)NN * HR * sizeof(float), stream);
    float* h_cur = hA;
    float* h_nxt = hB;
    for (int t = 0; t < TT; ++t) {
        // Gi = X_t @ W' + b'
        launch_gemm(inputd + (size_t)t * NN * FEAT, FEAT, Wp, G3H, 0,
                    Gi, G3H, NN, G3H, FEAT, bprime, stream);
        // Gh = h @ Whh^T + bhh
        launch_gemm(h_cur, HR, Whh, HR, 1, Gh, G3H, NN, G3H, HR, bhh, stream);
        gru_update_kernel<<<(NN * HR) / 256, 256, 0, stream>>>(Gi, Gh, h_cur, h_nxt);
        float* tmp = h_cur; h_cur = h_nxt; h_nxt = tmp;
    }
    const float* h = h_cur;

    // ---- 3) implicit-relation score vectors (fused GAT matvecs) ----
    fg_kernel<<<(NN + 255) / 256, 256, 0, stream>>>(h, v1, v2, fvec, gvec);

    // ---- 4) He = h @ W_exp ; S = He @ He^T into R ----
    launch_gemm(h, HR, W_exp, NHID, 0, He, NHID, NN, NHID, HR, nullptr, stream);
    launch_gemm(He, NHID, He, NHID, 1, R, NN, NN, NN, NHID, nullptr, stream);

    // ---- 5) relation = leaky(f_i+g_j) + A_Ind*S ; implicit -> d_out; mask -> R ----
    relation_kernel<<<(unsigned)(((size_t)NN * NN) / 256), 256, 0, stream>>>(
        fvec, gvec, A_Ind, R, out_imp);

    // ---- 6) row softmax (in place on R) ----
    softmax_row_kernel<<<NN, 256, 0, stream>>>(R);

    // ---- 7) gated message: gate_Mh = (h@graph_W) * sigmoid(h@W_att + b_att) ----
    launch_gemm(h, HR, graph_W, HS, 0, Mh, HS, NN, HS, HR, nullptr, stream);
    launch_gemm(h, HR, W_att, HS, 0, Att, HS, NN, HS, HR, b_att, stream);
    gate_kernel<<<(NN * HS) / 256, 256, 0, stream>>>(Mh, Att, gateMh);

    // ---- 8) H1 = A @ gate_Mh ----
    launch_gemm(R, NN, gateMh, HS, 0, Hs, HS, NN, HS, NN, nullptr, stream);

    // ---- 9) output head + regularization scalar ----
    head_kernel<<<(NN + 255) / 256, 256, 0, stream>>>(Hs, h, W1, b1, W2, b2, out_H);
    hipMemsetAsync(out_reg, 0, sizeof(float), stream);
}
